// SS2D_DSM_4956392259736
// MI455X (gfx1250) — matrix operand order repaired and checked by reading, not compiled, not run on hardware
//
#include <hip/hip_runtime.h>
#include <hip/hip_bf16.h>
#include <math.h>

#define DEV __device__ __forceinline__

constexpr int BATCH  = 4;
constexpr int DMODEL = 192;
constexpr int DINNER = 384;
constexpr int HH = 48;
constexpr int WW = 48;
constexpr int LL = HH * WW;             // 2304
constexpr int NDIR   = 4;
constexpr int NSTATE = 16;
constexpr int DTRANK = 12;
constexpr int XD = DTRANK + 2 * NSTATE; // 44
constexpr int NCOLS = BATCH * LL;       // 9216
constexpr int KB_IN  = DMODEL / 32;     // 6
constexpr int KB_DI  = DINNER / 32;     // 12

typedef __attribute__((ext_vector_type(16))) __bf16 v16bf;
typedef __attribute__((ext_vector_type(8)))  float  v8f;

union FragBF { unsigned short u[16]; uint4 q[2]; v16bf v; };

DEV unsigned short f2bf(float f) {
    unsigned int x = __float_as_uint(f);
    unsigned int r = (x + 0x7FFFu + ((x >> 16) & 1u)) >> 16;
    return (unsigned short)r;
}

// A fragment (16x32 bf16): lane holds M row; element runs K = h*16 .. h*16+15
// -> one contiguous 16x bf16 run in a row-major 32-wide K slice (16B aligned).
DEV void load_a_frag(FragBF& a, const unsigned short* __restrict__ Wrow, int kk, int half) {
    a.q[0] = *(const uint4*)(Wrow + kk + half * 16);
    a.q[1] = *(const uint4*)(Wrow + kk + half * 16 + 8);
}
// B fragment (32x16 bf16): lane = column, needs K = h*16 .. h*16+15 contiguous in staged
// layout [kblock][col][32]; base must already point at (...)*32 + half*16.
DEV void load_b_frag(FragBF& b, const unsigned short* __restrict__ base) {
    b.q[0] = *(const uint4*)(base);
    b.q[1] = *(const uint4*)(base + 8);
}
DEV void zero_frag(FragBF& a) {
    a.q[0] = make_uint4(0u, 0u, 0u, 0u);
    a.q[1] = make_uint4(0u, 0u, 0u, 0u);
}

DEV float gelu_exact(float x) {
    return 0.5f * x * (1.0f + erff(x * 0.70710678118654752f));
}
DEV float softplus_f(float x) {
    return (x > 20.f) ? x : log1pf(__expf(x));
}
// direction permutation: gather index for u AND merge index for y (same map)
DEV int map_idx(int kdir, int l) {
    int ll = (kdir >= 2) ? (LL - 1 - l) : l;
    return (kdir & 1) ? ((ll % HH) * WW + (ll / HH)) : ll;
}

// ---------------------------------------------------------------- elementwise
__global__ void cvt_bf16_kernel(const float* __restrict__ in,
                                unsigned short* __restrict__ out, int n) {
    int i = blockIdx.x * blockDim.x + threadIdx.x;
    if (i < n) out[i] = f2bf(in[i]);
}

__global__ void zero_kernel(float* __restrict__ p, int n) {
    int i = blockIdx.x * blockDim.x + threadIdx.x;
    if (i < n) p[i] = 0.f;
}

// stage x (B,192,L) f32 -> XS[(kb*NCOLS + n)*32 + t] bf16
__global__ void pack_x_kernel(const float* __restrict__ x,
                              unsigned short* __restrict__ XS) {
    int idx = blockIdx.x * blockDim.x + threadIdx.x;
    if (idx >= KB_IN * NCOLS * 32) return;
    int t  = idx & 31;
    int n  = (idx >> 5) % NCOLS;
    int kb = (idx >> 5) / NCOLS;
    int b = n / LL, l = n % LL;
    XS[idx] = f2bf(x[((size_t)b * DMODEL + kb * 32 + t) * LL + l]);
}

// pad dt_projs_weight (4,384,12) -> WS[(kd*384+m)*32 + k] (zeros for k>=12)
__global__ void pack_wdt_kernel(const float* __restrict__ W,
                                unsigned short* __restrict__ WS) {
    int idx = blockIdx.x * blockDim.x + threadIdx.x;
    if (idx >= NDIR * DINNER * 32) return;
    int k = idx & 31;
    int row = idx >> 5;
    WS[idx] = (k < DTRANK) ? f2bf(W[(size_t)row * DTRANK + k]) : (unsigned short)0;
}

// ------------------------------------------------- in_proj GEMM (768x192)x(192 x B*L)
__global__ void inproj_wmma_kernel(const unsigned short* __restrict__ Wbf,  // (768,192)
                                   const unsigned short* __restrict__ XS,   // staged
                                   float* __restrict__ xc_pre,
                                   float* __restrict__ z1) {
    int lane = threadIdx.x & 31, wid = threadIdx.x >> 5;
    int half = lane >> 4, lidx = lane & 15;
    int tm = blockIdx.x * 4 + wid;          // 0..47
    int n0 = blockIdx.y * 32 + lidx;        // two col tiles: n0, n0+16
    int m0 = tm * 16 + lidx;
    const unsigned short* Wrow = Wbf + (size_t)m0 * DMODEL;
    v8f acc0 = {}, acc1 = {};
    for (int kb = 0; kb < KB_IN; ++kb) {
        FragBF a, b0, b1;
        load_a_frag(a, Wrow, kb * 32, half);
        const unsigned short* bs = XS + (((size_t)kb * NCOLS + n0) * 32) + half * 16;
        load_b_frag(b0, bs);
        load_b_frag(b1, bs + 16 * 32);
        acc0 = __builtin_amdgcn_wmma_f32_16x16x32_bf16(false, a.v, false, b0.v,
                                                       (short)0, acc0, false, false);
        acc1 = __builtin_amdgcn_wmma_f32_16x16x32_bf16(false, a.v, false, b1.v,
                                                       (short)0, acc1, false, false);
    }
#pragma unroll
    for (int j = 0; j < 2; ++j) {
        int n = blockIdx.y * 32 + j * 16 + lidx;
        int b = n / LL, l = n % LL;
        v8f acc = j ? acc1 : acc0;
#pragma unroll
        for (int r = 0; r < 8; ++r) {
            int m = tm * 16 + r + 8 * half;
            float v = acc[r];
            if (m < DINNER) xc_pre[((size_t)b * DINNER + m) * LL + l] = v;
            else            z1[((size_t)b * DINNER + (m - DINNER)) * LL + l] = gelu_exact(v);
        }
    }
}

// ------------------------------------------------- depthwise 3x3 conv + bias + GELU
__global__ void dwconv_gelu_kernel(const float* __restrict__ xin,
                                   const float* __restrict__ cw,
                                   const float* __restrict__ cb,
                                   float* __restrict__ xout) {
    int idx = blockIdx.x * blockDim.x + threadIdx.x;
    if (idx >= BATCH * DINNER * LL) return;
    int l = idx % LL, d = (idx / LL) % DINNER, b = idx / (LL * DINNER);
    int h = l / WW, w = l % WW;
    float acc = cb[d];
#pragma unroll
    for (int kh = 0; kh < 3; ++kh) {
        int hh = h + kh - 1;
        if (hh < 0 || hh >= HH) continue;
#pragma unroll
        for (int kw = 0; kw < 3; ++kw) {
            int w2 = w + kw - 1;
            if (w2 < 0 || w2 >= WW) continue;
            acc += cw[d * 9 + kh * 3 + kw] * xin[((size_t)b * DINNER + d) * LL + hh * WW + w2];
        }
    }
    xout[idx] = gelu_exact(acc);
}

// ------------------------------------------------- CrossScan gather -> staged bf16 xs
// XSS[((bk*12 + kb)*LL + l)*32 + t]
__global__ void build_xs_kernel(const float* __restrict__ xc2,
                                unsigned short* __restrict__ XSS) {
    int idx = blockIdx.x * blockDim.x + threadIdx.x;
    if (idx >= BATCH * NDIR * KB_DI * LL * 32) return;
    int t  = idx & 31;
    int l  = (idx >> 5) % LL;
    int kb = ((idx >> 5) / LL) % KB_DI;
    int bk = ((idx >> 5) / LL) / KB_DI;
    int kd = bk % NDIR, b = bk / NDIR;
    XSS[idx] = f2bf(xc2[((size_t)b * DINNER + kb * 32 + t) * LL + map_idx(kd, l)]);
}

// ------------------------------------------------- x_proj GEMM (44x384) per (b,kdir)
__global__ void xproj_wmma_kernel(const unsigned short* __restrict__ Wxp,   // (4,44,384)
                                  const unsigned short* __restrict__ XSS,   // staged
                                  unsigned short* __restrict__ dtsS,        // (B*4, L, 32)
                                  float* __restrict__ BC) {                 // (B*4, 32, L)
    int lane = threadIdx.x & 31, wid = threadIdx.x >> 5;   // wid = tileM 0..2
    int half = lane >> 4, lidx = lane & 15;
    int bk = blockIdx.x;                   // 0..15
    int kd = bk % NDIR;
    int l  = blockIdx.y * 16 + lidx;
    int m0 = wid * 16 + lidx;              // 0..47
    v8f acc = {};
    for (int kb = 0; kb < KB_DI; ++kb) {
        FragBF a, bb;
        if (m0 < XD) load_a_frag(a, Wxp + ((size_t)kd * XD + m0) * DINNER, kb * 32, half);
        else         zero_frag(a);
        load_b_frag(bb, XSS + (((size_t)(bk * KB_DI + kb) * LL + l) * 32) + half * 16);
        acc = __builtin_amdgcn_wmma_f32_16x16x32_bf16(false, a.v, false, bb.v,
                                                      (short)0, acc, false, false);
    }
#pragma unroll
    for (int r = 0; r < 8; ++r) {
        int m = wid * 16 + r + 8 * half;
        float v = acc[r];
        if (m < 32)  // staged dts row: value for m<12, zero-pad for 12..31
            dtsS[((size_t)bk * LL + l) * 32 + m] = (m < DTRANK) ? f2bf(v) : (unsigned short)0;
        if (m >= DTRANK && m < XD)
            BC[((size_t)bk * 32 + (m - DTRANK)) * LL + l] = v;
    }
}

// ------------------------------------------------- dt_proj GEMM (384x12) + bias + softplus
__global__ void dtproj_wmma_kernel(const unsigned short* __restrict__ WdtS,  // (4*384, 32) padded
                                   const unsigned short* __restrict__ dtsS,  // (B*4, L, 32)
                                   const float* __restrict__ dt_bias,        // (4,384)
                                   float* __restrict__ delta) {              // (B*4,384,L)
    int lane = threadIdx.x & 31, wid = threadIdx.x >> 5;
    int half = lane >> 4, lidx = lane & 15;
    int bk = blockIdx.x / 6;
    int tm = (blockIdx.x % 6) * 4 + wid;   // 0..23
    int kd = bk % NDIR;
    int l  = blockIdx.y * 16 + lidx;
    int m0 = tm * 16 + lidx;
    FragBF a, bb;
    load_a_frag(a, WdtS + (size_t)(kd * DINNER + m0) * 32, 0, half);
    load_b_frag(bb, dtsS + (((size_t)bk * LL + l) * 32) + half * 16);
    v8f acc = {};
    acc = __builtin_amdgcn_wmma_f32_16x16x32_bf16(false, a.v, false, bb.v,
                                                  (short)0, acc, false, false);
#pragma unroll
    for (int r = 0; r < 8; ++r) {
        int m = tm * 16 + r + 8 * half;
        delta[((size_t)bk * DINNER + m) * LL + l] = softplus_f(acc[r] + dt_bias[kd * DINNER + m]);
    }
}

// ------------------------------------------------- selective scan + fused CrossMerge
// 16 lanes per (b,k,d) row (one per state n); wave32 = 2 rows.
__global__ void scan_kernel(const float* __restrict__ xc2,
                            const float* __restrict__ delta,
                            const float* __restrict__ BC,
                            const float* __restrict__ A_logs,
                            const float* __restrict__ Ds,
                            float* __restrict__ ymerge) {
    int gtid = blockIdx.x * blockDim.x + threadIdx.x;
    int wave = gtid >> 5;
    int lane = threadIdx.x & 31;
    int n    = lane & 15;
    int sub  = lane >> 4;
    int gd   = wave * 2 + sub;             // flat (b,kdir,d)
    if (gd >= BATCH * NDIR * DINNER) return;
    int d  = gd % DINNER;
    int kd = (gd / DINNER) % NDIR;
    int b  = gd / (DINNER * NDIR);
    float Aval = -__expf(A_logs[((size_t)kd * DINNER + d) * NSTATE + n]);
    float Dval = Ds[kd * DINNER + d];
    const float* drow = delta + (size_t)gd * LL;
    const float* urow = xc2 + ((size_t)b * DINNER + d) * LL;
    const float* Brow = BC + ((size_t)(b * NDIR + kd) * 32 + n) * LL;
    const float* Crow = Brow + (size_t)NSTATE * LL;
    float* yrow = ymerge + ((size_t)b * DINNER + d) * LL;
    float h = 0.f;
    for (int l = 0; l < LL; ++l) {
        // latency hiding for the serial dependence chain: pull lines ~1KB ahead
        if ((l & 31) == 0) {
            int lp = l + 256;
            if (lp < LL) {
                __builtin_prefetch(drow + lp, 0, 0);
                __builtin_prefetch(Brow + lp, 0, 0);
                __builtin_prefetch(Crow + lp, 0, 0);
                __builtin_prefetch(urow + map_idx(kd, lp), 0, 0);
            }
        }
        int gi = map_idx(kd, l);
        float u  = urow[gi];
        float dl = drow[l];
        h = h * __expf(dl * Aval) + (dl * u) * Brow[l];
        float y = h * Crow[l];
        y += __shfl_xor(y, 1);
        y += __shfl_xor(y, 2);
        y += __shfl_xor(y, 4);
        y += __shfl_xor(y, 8);
        if (n == 0) atomicAdd(&yrow[gi], y + u * Dval);
    }
}

// ------------------------------------------------- LayerNorm(channel) + gate -> staged bf16
// YS[((b*12 + d/32)*LL + l)*32 + d%32]
__global__ void ln_gate_kernel(const float* __restrict__ ymerge,
                               const float* __restrict__ z1,
                               const float* __restrict__ lnw,
                               const float* __restrict__ lnb,
                               unsigned short* __restrict__ YS) {
    __shared__ float red[128];
    int bl = blockIdx.x;
    int b = bl / LL, l = bl % LL;
    int tid = threadIdx.x;
    float vals[3];
    float s = 0.f;
#pragma unroll
    for (int i = 0; i < 3; ++i) {
        int d = tid + i * 128;
        vals[i] = ymerge[((size_t)b * DINNER + d) * LL + l];
        s += vals[i];
    }
    red[tid] = s; __syncthreads();
    for (int st = 64; st > 0; st >>= 1) { if (tid < st) red[tid] += red[tid + st]; __syncthreads(); }
    float mu = red[0] * (1.f / DINNER);
    __syncthreads();
    float vs = 0.f;
#pragma unroll
    for (int i = 0; i < 3; ++i) { float t = vals[i] - mu; vs += t * t; }
    red[tid] = vs; __syncthreads();
    for (int st = 64; st > 0; st >>= 1) { if (tid < st) red[tid] += red[tid + st]; __syncthreads(); }
    float rstd = rsqrtf(red[0] * (1.f / DINNER) + 1e-5f);
#pragma unroll
    for (int i = 0; i < 3; ++i) {
        int d = tid + i * 128;
        float v = (vals[i] - mu) * rstd * lnw[d] + lnb[d];
        v *= z1[((size_t)b * DINNER + d) * LL + l];
        YS[(((size_t)b * KB_DI + (d >> 5)) * LL + l) * 32 + (d & 31)] = f2bf(v);
    }
}

// ------------------------------------------------- out_proj GEMM (192x384)x(384 x B*L)
__global__ void outproj_wmma_kernel(const unsigned short* __restrict__ Wbf,  // (192,384)
                                    const unsigned short* __restrict__ YS,   // staged
                                    float* __restrict__ out) {               // (B,192,L)
    int lane = threadIdx.x & 31, wid = threadIdx.x >> 5;
    int half = lane >> 4, lidx = lane & 15;
    int tm = blockIdx.x * 4 + wid;          // 0..11
    int n0 = blockIdx.y * 32 + lidx;        // two col tiles (never cross batch: L%32==0)
    int m0 = tm * 16 + lidx;
    int b = n0 / LL, l = n0 % LL;
    const unsigned short* Wrow = Wbf + (size_t)m0 * DINNER;
    v8f acc0 = {}, acc1 = {};
    for (int kb = 0; kb < KB_DI; ++kb) {
        FragBF a, b0, b1;
        load_a_frag(a, Wrow, kb * 32, half);
        const unsigned short* bs = YS + (((size_t)(b * KB_DI + kb) * LL + l) * 32) + half * 16;
        load_b_frag(b0, bs);
        load_b_frag(b1, bs + 16 * 32);
        acc0 = __builtin_amdgcn_wmma_f32_16x16x32_bf16(false, a.v, false, b0.v,
                                                       (short)0, acc0, false, false);
        acc1 = __builtin_amdgcn_wmma_f32_16x16x32_bf16(false, a.v, false, b1.v,
                                                       (short)0, acc1, false, false);
    }
#pragma unroll
    for (int j = 0; j < 2; ++j) {
        int lj = l + j * 16;
        v8f acc = j ? acc1 : acc0;
#pragma unroll
        for (int r = 0; r < 8; ++r) {
            int m = tm * 16 + r + 8 * half;
            out[((size_t)b * DMODEL + m) * LL + lj] = acc[r];
        }
    }
}

// ----------------------------------------------------------------------------
extern "C" void kernel_launch(void* const* d_in, const int* in_sizes, int n_in,
                              void* d_out, int out_size, void* d_ws, size_t ws_size,
                              hipStream_t stream) {
    (void)in_sizes; (void)n_in; (void)out_size; (void)ws_size;
    const float* x         = (const float*)d_in[0];
    const float* in_proj_w = (const float*)d_in[1];
    const float* conv_w    = (const float*)d_in[2];
    const float* conv_b    = (const float*)d_in[3];
    const float* x_proj_w  = (const float*)d_in[4];
    const float* dt_w      = (const float*)d_in[5];
    const float* dt_b      = (const float*)d_in[6];
    const float* A_logs    = (const float*)d_in[7];
    const float* Ds        = (const float*)d_in[8];
    const float* ln_w      = (const float*)d_in[9];
    const float* ln_b      = (const float*)d_in[10];
    const float* out_w     = (const float*)d_in[11];
    float* out = (float*)d_out;

    char* ws = (char*)d_ws;
    size_t off = 0;
    auto alloc = [&](size_t bytes) -> void* {
        void* p = ws + off;
        off += (bytes + 255) & ~(size_t)255;
        return p;
    };
    const size_t SZ1 = (size_t)BATCH * DINNER * LL;          // 3,538,944
    const size_t SZK = (size_t)BATCH * NDIR * DINNER * LL;   // 14,155,776

    float* xc_pre = (float*)alloc(SZ1 * 4);
    float* z1     = (float*)alloc(SZ1 * 4);
    float* xc2    = (float*)alloc(SZ1 * 4);
    float* ymerge = (float*)alloc(SZ1 * 4);
    float* delta  = (float*)alloc(SZK * 4);
    float* BCbuf  = (float*)alloc((size_t)BATCH * NDIR * 32 * LL * 4);
    unsigned short* XSS   = (unsigned short*)alloc(SZK * 2);                          // staged xs
    unsigned short* dtsS  = (unsigned short*)alloc((size_t)BATCH * NDIR * LL * 32 * 2);
    unsigned short* YS    = (unsigned short*)alloc(SZ1 * 2);                          // staged gated y
    unsigned short* XS    = (unsigned short*)alloc((size_t)KB_IN * NCOLS * 32 * 2);   // staged x
    unsigned short* Winbf = (unsigned short*)alloc((size_t)2 * DINNER * DMODEL * 2);
    unsigned short* Wxpbf = (unsigned short*)alloc((size_t)NDIR * XD * DINNER * 2);
    unsigned short* WdtS  = (unsigned short*)alloc((size_t)NDIR * DINNER * 32 * 2);
    unsigned short* Woutbf= (unsigned short*)alloc((size_t)DMODEL * DINNER * 2);

    auto cvt = [&](const float* src, unsigned short* dst, int n) {
        cvt_bf16_kernel<<<(n + 255) / 256, 256, 0, stream>>>(src, dst, n);
    };
    pack_x_kernel<<<(KB_IN * NCOLS * 32 + 255) / 256, 256, 0, stream>>>(x, XS);
    cvt(in_proj_w, Winbf,  2 * DINNER * DMODEL);
    cvt(x_proj_w,  Wxpbf,  NDIR * XD * DINNER);
    pack_wdt_kernel<<<(NDIR * DINNER * 32 + 255) / 256, 256, 0, stream>>>(dt_w, WdtS);
    cvt(out_w,     Woutbf, DMODEL * DINNER);

    // in_proj: 48 row-tiles x 576 col-tiles (2 col-tiles per wave)
    inproj_wmma_kernel<<<dim3(12, 288), 128, 0, stream>>>(Winbf, XS, xc_pre, z1);
    dwconv_gelu_kernel<<<(int)((SZ1 + 255) / 256), 256, 0, stream>>>(xc_pre, conv_w, conv_b, xc2);
    build_xs_kernel<<<(int)((SZK + 255) / 256), 256, 0, stream>>>(xc2, XSS);
    // x_proj: (b,kdir)=16 blocks x 3 waves(tileM); 144 col-tiles
    xproj_wmma_kernel<<<dim3(16, 144), 96, 0, stream>>>(Wxpbf, XSS, dtsS, BCbuf);
    // dt_proj: 24 row-tiles per (b,kdir): grid.x = 16*6, 4 waves each
    dtproj_wmma_kernel<<<dim3(96, 144), 128, 0, stream>>>(WdtS, dtsS, dt_b, delta);
    zero_kernel<<<(int)((SZ1 + 255) / 256), 256, 0, stream>>>(ymerge, (int)SZ1);
    // scan: 6144 rows / 2 per wave = 3072 waves = 384 blocks of 256
    scan_kernel<<<384, 256, 0, stream>>>(xc2, delta, BCbuf, A_logs, Ds, ymerge);
    ln_gate_kernel<<<BATCH * LL, 128, 0, stream>>>(ymerge, z1, ln_w, ln_b, YS);
    // out_proj: 12 row-tiles x 576 col-tiles (2 per wave)
    outproj_wmma_kernel<<<dim3(3, 288), 128, 0, stream>>>(Woutbf, YS, out);
}
